// OptimizedGeometricRound_8186207666911
// MI455X (gfx1250) — matrix operand order repaired and checked by reading, not compiled, not run on hardware
//
#include <hip/hip_runtime.h>

typedef __attribute__((ext_vector_type(16))) __bf16 v16bf;
typedef __attribute__((ext_vector_type(8)))  float  v8f;
typedef __attribute__((ext_vector_type(4)))  int    v4i;

union FragU { uint4 u[2]; v16bf v; };

#ifndef __has_builtin
#define __has_builtin(x) 0
#endif

#if __has_builtin(__builtin_amdgcn_global_load_async_to_lds_b128)
#define HAVE_ASYNC_COPY 1
#else
#define HAVE_ASYNC_COPY 0
#endif

static __device__ __forceinline__ void wait_async0() {
#if __has_builtin(__builtin_amdgcn_s_wait_asynccnt)
  __builtin_amdgcn_s_wait_asynccnt(0);
#else
  asm volatile("s_wait_asynccnt 0" ::: "memory");
#endif
}

#if HAVE_ASYNC_COPY
static __device__ __forceinline__ void async_cp_b128(const void* g, void* l) {
  __builtin_amdgcn_global_load_async_to_lds_b128(
      (__attribute__((address_space(1))) v4i*)g,
      (__attribute__((address_space(3))) v4i*)l, 0, 0);
}
#endif

static __device__ __forceinline__ unsigned short f2bf(float f) {
  unsigned int u = __float_as_uint(f);
  u += 0x7FFFu + ((u >> 16) & 1u);          // round-to-nearest-even
  return (unsigned short)(u >> 16);
}

#define WMMA_BF16(a, b, c) \
  __builtin_amdgcn_wmma_f32_16x16x32_bf16(false, (a), false, (b), (short)0, (c), false, false)

// ---------------------------------------------------------------- cvt f32->bf16
__global__ __launch_bounds__(256)
void cvt_f32_bf16(const float* __restrict__ src, unsigned short* __restrict__ dst, int n) {
  int i = (blockIdx.x * 256 + threadIdx.x) * 4;
  if (i + 3 < n) {
    float4 f = *(const float4*)(src + i);
    ushort4 o;
    o.x = f2bf(f.x); o.y = f2bf(f.y); o.z = f2bf(f.z); o.w = f2bf(f.w);
    *(ushort4*)(dst + i) = o;
  }
}

// ---------------------------------------------------------------- geo product + mix + LN
// Cl(3,0) Cayley tables: basis 1,e1,e2,e3,e12,e13,e23,e123 ; e_i^2=+1
constexpr int GK[64] = {
  0,1,2,3,4,5,6,7,
  1,0,4,5,2,3,7,6,
  2,4,0,6,1,7,3,5,
  3,5,6,0,7,1,2,4,
  4,2,1,7,0,6,5,3,
  5,3,7,1,6,0,4,2,
  6,7,3,2,5,4,0,1,
  7,6,5,4,3,2,1,0};
constexpr float GSgn[64] = {
  1, 1, 1, 1, 1, 1, 1, 1,
  1, 1, 1, 1, 1, 1, 1, 1,
  1,-1, 1, 1,-1,-1, 1,-1,
  1,-1,-1, 1, 1,-1,-1, 1,
  1,-1, 1, 1,-1,-1, 1,-1,
  1,-1,-1, 1, 1,-1,-1, 1,
  1, 1,-1, 1,-1, 1,-1,-1,
  1, 1,-1, 1,-1, 1,-1,-1};

__global__ __launch_bounds__(256)
void geo_ln_kernel(const float* __restrict__ x, const float* __restrict__ iw,
                   const float* __restrict__ gg, const float* __restrict__ lnw,
                   const float* __restrict__ lnb, unsigned short* __restrict__ flat) {
  __shared__ float smix[8][1024];
  __shared__ float wred[2][8][8];   // [sum|sumsq][wave][blade]
  __shared__ float smu[8], srs[8];
  const int t = threadIdx.x;
  const int n = blockIdx.x;
  const float g   = 1.f / (1.f + __expf(-gg[0]));
  const float omg = 1.f - g;

  float c[64];
#pragma unroll
  for (int p = 0; p < 64; ++p) c[p] = GSgn[p] / (1.f + __expf(-iw[p]));

  float s[8], q[8];
#pragma unroll
  for (int k = 0; k < 8; ++k) { s[k] = 0.f; q[k] = 0.f; }

  const float* xb = x + ((size_t)n << 13);   // n*8*1024
#pragma unroll
  for (int cc = 0; cc < 4; ++cc) {
    const int d = cc * 256 + t;
    float xv[8];
#pragma unroll
    for (int i = 0; i < 8; ++i) xv[i] = xb[(i << 10) + d];
    float geo[8];
#pragma unroll
    for (int k = 0; k < 8; ++k) geo[k] = 0.f;
#pragma unroll
    for (int i = 0; i < 8; ++i)
#pragma unroll
      for (int j = 0; j < 8; ++j)
        geo[GK[i * 8 + j]] += c[i * 8 + j] * xv[i] * xv[j];
#pragma unroll
    for (int k = 0; k < 8; ++k) {
      float m = g * geo[k] + omg * xv[k];
      smix[k][d] = m;
      s[k] += m; q[k] += m * m;
    }
  }
  // wave-level reduction (wave32)
#pragma unroll
  for (int k = 0; k < 8; ++k)
#pragma unroll
    for (int off = 16; off > 0; off >>= 1) {
      s[k] += __shfl_xor(s[k], off, 32);
      q[k] += __shfl_xor(q[k], off, 32);
    }
  const int lane = t & 31, wv = t >> 5;
  if (lane == 0) {
#pragma unroll
    for (int k = 0; k < 8; ++k) { wred[0][wv][k] = s[k]; wred[1][wv][k] = q[k]; }
  }
  __syncthreads();
  if (t < 8) {
    float S = 0.f, Q = 0.f;
#pragma unroll
    for (int w = 0; w < 8; ++w) { S += wred[0][w][t]; Q += wred[1][w][t]; }
    float mu  = S * (1.f / 1024.f);
    float var = Q * (1.f / 1024.f) - mu * mu;
    smu[t] = mu;
    srs[t] = rsqrtf(var + 1e-5f);
  }
  __syncthreads();
  unsigned short* fb = flat + ((size_t)n << 13);
#pragma unroll
  for (int cc = 0; cc < 4; ++cc) {
    const int d = cc * 256 + t;
    const float lw = lnw[d], lb = lnb[d];
#pragma unroll
    for (int k = 0; k < 8; ++k) {
      float v = (smix[k][d] - smu[k]) * srs[k] * lw + lb;
      fb[(k << 10) + d] = f2bf(v);
    }
  }
}

// ---------------------------------------------------------------- GEMM1: h = silu(A*Wg) * (A*Wu)
// A: 32768x1024 bf16 ; Wg,Wu: 1024x4096 bf16 ; H: 32768x4096 bf16
__global__ __launch_bounds__(256, 1)
void gemm_swiglu_kernel(const unsigned short* __restrict__ A,
                        const unsigned short* __restrict__ Bg,
                        const unsigned short* __restrict__ Bu,
                        unsigned short* __restrict__ H) {
  constexpr int K = 1024, N = 4096;
  constexpr int BM = 128, BN = 64, BK = 32, NK = K / BK;
  constexpr int LDA = 56, LDB = 72;   // padded, 16B-aligned rows, conflict-free b128
  __shared__ __align__(16) unsigned short sA [2][BM * LDA];
  __shared__ __align__(16) unsigned short sBg[2][BK * LDB];
  __shared__ __align__(16) unsigned short sBu[2][BK * LDB];

  const int t = threadIdx.x;
  const int lane = t & 31, wv = t >> 5;
  const int hf = lane >> 4, l15 = lane & 15;
  const int bm = (int)(blockIdx.x / (N / BN));
  const int bn = (int)(blockIdx.x % (N / BN));
  const int m0 = bm * BM, n0 = bn * BN;
  const int wM = (wv >> 1) * 32, wN = (wv & 1) * 32;

  const int brow = t >> 3, bcv = t & 7;

#if HAVE_ASYNC_COPY
  auto stage = [&](int kt, int buf) {
#pragma unroll
    for (int u = 0; u < 2; ++u) {
      int idx = t + u * 256;
      int row = idx >> 2, cv = idx & 3;
      async_cp_b128(A + (size_t)(m0 + row) * K + kt * BK + cv * 8,
                    &sA[buf][row * LDA + cv * 8]);
    }
    size_t off = (size_t)(kt * BK + brow) * N + n0 + bcv * 8;
    async_cp_b128(Bg + off, &sBg[buf][brow * LDB + bcv * 8]);
    async_cp_b128(Bu + off, &sBu[buf][brow * LDB + bcv * 8]);
  };
#else
  uint4 ra[2], rbg, rbu;
  auto gload = [&](int kt) {
#pragma unroll
    for (int u = 0; u < 2; ++u) {
      int idx = t + u * 256;
      int row = idx >> 2, cv = idx & 3;
      ra[u] = *(const uint4*)(A + (size_t)(m0 + row) * K + kt * BK + cv * 8);
    }
    size_t off = (size_t)(kt * BK + brow) * N + n0 + bcv * 8;
    rbg = *(const uint4*)(Bg + off);
    rbu = *(const uint4*)(Bu + off);
  };
  auto swrite = [&](int buf) {
#pragma unroll
    for (int u = 0; u < 2; ++u) {
      int idx = t + u * 256;
      int row = idx >> 2, cv = idx & 3;
      *(uint4*)(&sA[buf][row * LDA + cv * 8]) = ra[u];
    }
    *(uint4*)(&sBg[buf][brow * LDB + bcv * 8]) = rbg;
    *(uint4*)(&sBu[buf][brow * LDB + bcv * 8]) = rbu;
  };
#endif

  v8f zero = {0.f, 0.f, 0.f, 0.f, 0.f, 0.f, 0.f, 0.f};
  v8f accg[2][2], accu[2][2];
#pragma unroll
  for (int i = 0; i < 2; ++i)
#pragma unroll
    for (int j = 0; j < 2; ++j) { accg[i][j] = zero; accu[i][j] = zero; }

#if HAVE_ASYNC_COPY
  stage(0, 0);
  wait_async0();
  __syncthreads();
#else
  gload(0); swrite(0);
  __syncthreads();
#endif

  for (int kt = 0; kt < NK; ++kt) {
    const int cur = kt & 1;
#if HAVE_ASYNC_COPY
    if (kt + 1 < NK) stage(kt + 1, cur ^ 1);
#else
    if (kt + 1 < NK) gload(kt + 1);
#endif

    // distinct fragment registers for gate and up B operands: no WAR chains
    FragU a0, a1, bg0, bg1, bu0, bu1;
    const unsigned short* pa = &sA[cur][0];
    a0.u[0] = *(const uint4*)(pa + (wM + l15) * LDA + hf * 8);
    a0.u[1] = *(const uint4*)(pa + (wM + l15) * LDA + 16 + hf * 8);
    a1.u[0] = *(const uint4*)(pa + (wM + 16 + l15) * LDA + hf * 8);
    a1.u[1] = *(const uint4*)(pa + (wM + 16 + l15) * LDA + 16 + hf * 8);
    auto ldb = [&](FragU& f, const unsigned short* p, int col) {
      unsigned int w[8];
#pragma unroll
      for (int j = 0; j < 8; ++j) {
        int k = (j < 4) ? hf * 8 + 2 * j : 16 + hf * 8 + 2 * (j - 4);
        w[j] = (unsigned int)p[k * LDB + col] | ((unsigned int)p[(k + 1) * LDB + col] << 16);
      }
      f.u[0] = make_uint4(w[0], w[1], w[2], w[3]);
      f.u[1] = make_uint4(w[4], w[5], w[6], w[7]);
    };

    const unsigned short* pg = &sBg[cur][0];
    const unsigned short* pu = &sBu[cur][0];
    ldb(bg0, pg, wN + l15);      ldb(bg1, pg, wN + 16 + l15);
    ldb(bu0, pu, wN + l15);      ldb(bu1, pu, wN + 16 + l15);

    accg[0][0] = WMMA_BF16(a0.v, bg0.v, accg[0][0]);
    accg[0][1] = WMMA_BF16(a0.v, bg1.v, accg[0][1]);
    accg[1][0] = WMMA_BF16(a1.v, bg0.v, accg[1][0]);
    accg[1][1] = WMMA_BF16(a1.v, bg1.v, accg[1][1]);
    accu[0][0] = WMMA_BF16(a0.v, bu0.v, accu[0][0]);
    accu[0][1] = WMMA_BF16(a0.v, bu1.v, accu[0][1]);
    accu[1][0] = WMMA_BF16(a1.v, bu0.v, accu[1][0]);
    accu[1][1] = WMMA_BF16(a1.v, bu1.v, accu[1][1]);

    if (kt + 1 < NK) {
#if HAVE_ASYNC_COPY
      wait_async0();
#else
      swrite(cur ^ 1);
#endif
      __syncthreads();
    }
  }

#pragma unroll
  for (int mi = 0; mi < 2; ++mi)
#pragma unroll
    for (int ni = 0; ni < 2; ++ni)
#pragma unroll
      for (int r = 0; r < 8; ++r) {
        float gv = accg[mi][ni][r];
        float uv = accu[mi][ni][r];
        float hv = gv / (1.f + __expf(-gv)) * uv;   // silu(g)*u
        int row = m0 + wM + mi * 16 + 8 * hf + r;
        int col = n0 + wN + ni * 16 + l15;
        H[(size_t)row * N + col] = f2bf(hv);
      }
}

// ---------------------------------------------------------------- GEMM2: out = x + H * Wd
// H: 32768x4096 bf16 ; Wd: 4096x1024 bf16 ; out fp32 32768x1024
__global__ __launch_bounds__(256, 1)
void gemm_down_kernel(const unsigned short* __restrict__ A,
                      const unsigned short* __restrict__ B,
                      const float* __restrict__ X,
                      float* __restrict__ O) {
  constexpr int K = 4096, N = 1024;
  constexpr int BM = 128, BN = 128, BK = 32, NK = K / BK;
  constexpr int LDA = 56, LDB = 136;
  __shared__ __align__(16) unsigned short sA[2][BM * LDA];
  __shared__ __align__(16) unsigned short sB[2][BK * LDB];

  const int t = threadIdx.x;
  const int lane = t & 31, wv = t >> 5;
  const int hf = lane >> 4, l15 = lane & 15;
  const int bm = (int)(blockIdx.x / (N / BN));
  const int bn = (int)(blockIdx.x % (N / BN));
  const int m0 = bm * BM, n0 = bn * BN;
  const int wM = (wv >> 2) * 64, wN = (wv & 3) * 32;

#if HAVE_ASYNC_COPY
  auto stage = [&](int kt, int buf) {
#pragma unroll
    for (int u = 0; u < 2; ++u) {
      int idx = t + u * 256;
      int arow = idx >> 2, acv = idx & 3;
      async_cp_b128(A + (size_t)(m0 + arow) * K + kt * BK + acv * 8,
                    &sA[buf][arow * LDA + acv * 8]);
      int brow = idx >> 4, bcv = idx & 15;
      async_cp_b128(B + (size_t)(kt * BK + brow) * N + n0 + bcv * 8,
                    &sB[buf][brow * LDB + bcv * 8]);
    }
  };
#else
  uint4 ra[2], rb[2];
  auto gload = [&](int kt) {
#pragma unroll
    for (int u = 0; u < 2; ++u) {
      int idx = t + u * 256;
      int arow = idx >> 2, acv = idx & 3;
      ra[u] = *(const uint4*)(A + (size_t)(m0 + arow) * K + kt * BK + acv * 8);
      int brow = idx >> 4, bcv = idx & 15;
      rb[u] = *(const uint4*)(B + (size_t)(kt * BK + brow) * N + n0 + bcv * 8);
    }
  };
  auto swrite = [&](int buf) {
#pragma unroll
    for (int u = 0; u < 2; ++u) {
      int idx = t + u * 256;
      int arow = idx >> 2, acv = idx & 3;
      *(uint4*)(&sA[buf][arow * LDA + acv * 8]) = ra[u];
      int brow = idx >> 4, bcv = idx & 15;
      *(uint4*)(&sB[buf][brow * LDB + bcv * 8]) = rb[u];
    }
  };
#endif

  v8f zero = {0.f, 0.f, 0.f, 0.f, 0.f, 0.f, 0.f, 0.f};
  v8f acc[4][2];
#pragma unroll
  for (int i = 0; i < 4; ++i)
#pragma unroll
    for (int j = 0; j < 2; ++j) acc[i][j] = zero;

#if HAVE_ASYNC_COPY
  stage(0, 0);
  wait_async0();
  __syncthreads();
#else
  gload(0); swrite(0);
  __syncthreads();
#endif

  for (int kt = 0; kt < NK; ++kt) {
    const int cur = kt & 1;
#if HAVE_ASYNC_COPY
    if (kt + 1 < NK) stage(kt + 1, cur ^ 1);
#else
    if (kt + 1 < NK) gload(kt + 1);
#endif

    FragU a[4], b[2];
    const unsigned short* pa = &sA[cur][0];
#pragma unroll
    for (int mi = 0; mi < 4; ++mi) {
      a[mi].u[0] = *(const uint4*)(pa + (wM + mi * 16 + l15) * LDA + hf * 8);
      a[mi].u[1] = *(const uint4*)(pa + (wM + mi * 16 + l15) * LDA + 16 + hf * 8);
    }
    const unsigned short* pb = &sB[cur][0];
#pragma unroll
    for (int ni = 0; ni < 2; ++ni) {
      const int col = wN + ni * 16 + l15;
      unsigned int w[8];
#pragma unroll
      for (int j = 0; j < 8; ++j) {
        int k = (j < 4) ? hf * 8 + 2 * j : 16 + hf * 8 + 2 * (j - 4);
        w[j] = (unsigned int)pb[k * LDB + col] | ((unsigned int)pb[(k + 1) * LDB + col] << 16);
      }
      b[ni].u[0] = make_uint4(w[0], w[1], w[2], w[3]);
      b[ni].u[1] = make_uint4(w[4], w[5], w[6], w[7]);
    }
#pragma unroll
    for (int mi = 0; mi < 4; ++mi)
#pragma unroll
      for (int ni = 0; ni < 2; ++ni)
        acc[mi][ni] = WMMA_BF16(a[mi].v, b[ni].v, acc[mi][ni]);

    if (kt + 1 < NK) {
#if HAVE_ASYNC_COPY
      wait_async0();
#else
      swrite(cur ^ 1);
#endif
      __syncthreads();
    }
  }

#pragma unroll
  for (int mi = 0; mi < 4; ++mi)
#pragma unroll
    for (int ni = 0; ni < 2; ++ni)
#pragma unroll
      for (int r = 0; r < 8; ++r) {
        int row = m0 + wM + mi * 16 + 8 * hf + r;
        int col = n0 + wN + ni * 16 + l15;
        size_t off = (size_t)row * N + col;
        O[off] = X[off] + acc[mi][ni][r];
      }
}

// ---------------------------------------------------------------- launch
extern "C" void kernel_launch(void* const* d_in, const int* in_sizes, int n_in,
                              void* d_out, int out_size, void* d_ws, size_t ws_size,
                              hipStream_t stream) {
  const float* x   = (const float*)d_in[0];
  const float* iw  = (const float*)d_in[1];
  const float* gg  = (const float*)d_in[2];
  const float* lnw = (const float*)d_in[3];
  const float* lnb = (const float*)d_in[4];
  const float* wg  = (const float*)d_in[5];
  const float* wu  = (const float*)d_in[6];
  const float* wd  = (const float*)d_in[7];

  const size_t M = 32768;          // 4096 tokens * 8 blades
  unsigned short* flat = (unsigned short*)d_ws;               // M x 1024 bf16
  unsigned short* h    = flat + M * 1024;                     // M x 4096 bf16
  unsigned short* wgb  = h + M * 4096;                        // 1024 x 4096 bf16
  unsigned short* wub  = wgb + (size_t)1024 * 4096;
  unsigned short* wdb  = wub + (size_t)1024 * 4096;

  const int wn = 1024 * 4096;
  cvt_f32_bf16<<<wn / 1024, 256, 0, stream>>>(wg, wgb, wn);
  cvt_f32_bf16<<<wn / 1024, 256, 0, stream>>>(wu, wub, wn);
  cvt_f32_bf16<<<wn / 1024, 256, 0, stream>>>(wd, wdb, wn);

  geo_ln_kernel<<<4096, 256, 0, stream>>>(x, iw, gg, lnw, lnb, flat);

  gemm_swiglu_kernel<<<(int)((M / 128) * (4096 / 64)), 256, 0, stream>>>(flat, wgb, wub, h);

  gemm_down_kernel<<<(int)((M / 128) * (1024 / 128)), 256, 0, stream>>>(h, wdb, x, (float*)d_out);
}
